// MeshGaussiansField_1589137899547
// MI455X (gfx1250) — matrix operand order repaired and checked by reading, not compiled, not run on hardware
//
#include <hip/hip_runtime.h>

typedef __attribute__((ext_vector_type(16))) __bf16        v16bf;
typedef __attribute__((ext_vector_type(8)))  float         v8f;
typedef __attribute__((ext_vector_type(4)))  unsigned int  u32x4;

#define C0F   0.28209479177387814f
#define ASTR  256          // row stride (elems) of 256-wide activation buffer
#define XSTR  288          // row stride (elems) of padded input buffer (265->288)

// dword offsets inside prepped-weight workspace (each tile = 256 dwords = 1KB)
#define GEO_DW    0            // geo_w1[:,1:257]  : 8 kt x 16 nt = 128 tiles
#define R0_DW     (128*256)    // rw0 (265->288)   : 9 kt x 16 nt = 144 tiles
#define R1_DW     (272*256)    // rw1              : 128 tiles
#define R2_DW     (400*256)
#define R3_DW     (528*256)
#define HEAD_DW   (656*256)    // [rw4|sw|aw|0pad] 256x16 : 8 tiles
#define GEOOP_DW  (664*256)    // geo_w1[:,0] in col0, 256x16 : 8 tiles
#define GW0_DW    (672*256)    // geo_w0 (3->32 pad) x 256 : 16 tiles (KT=1)
#define NPREP_TILES 688

__device__ __forceinline__ unsigned short f2bf(float f) {
  unsigned u = __float_as_uint(f);
  u += 0x7FFFu + ((u >> 16) & 1u);          // round-to-nearest-even
  return (unsigned short)(u >> 16);
}
__device__ __forceinline__ float bf2f(unsigned short h) {
  return __uint_as_float(((unsigned)h) << 16);
}
__device__ __forceinline__ unsigned pack2(float lo, float hi) {
  return (unsigned)f2bf(lo) | ((unsigned)f2bf(hi) << 16);
}
__device__ __forceinline__ v16bf mkv(u32x4 a, u32x4 b) {
  union { u32x4 q[2]; v16bf v; } u;
  u.q[0] = a; u.q[1] = b; return u.v;
}
// CDNA5 split dep-counter wait: orders wave-private LDS store->load across lanes
__device__ __forceinline__ void lds_fence() {
  __asm__ volatile("s_wait_dscnt 0" ::: "memory");
}
__device__ __forceinline__ float sigmoidf(float x) { return 1.0f / (1.0f + __expf(-x)); }
// branchless, overflow-safe: log1p(exp(x)) == max(x,0) + log(1 + exp(-|x|))
__device__ __forceinline__ float softplusf(float x) {
  return fmaxf(x, 0.0f) + __logf(1.0f + __expf(-fabsf(x)));
}

// ---------------------------------------------------------------------------
// Weight prep: f32 row-major -> bf16 WMMA-B tiles (1KB/tile, 2x b128 per lane)
// B layout per tile (32x16, K x N): lane n=lane%16, khalf=lane/16,
//   dword v holds K = kt*32 + khalf*16 + {2v, 2v+1}
// ---------------------------------------------------------------------------
__global__ __launch_bounds__(32) void prep_weights(
    const float* __restrict__ geo_w0, const float* __restrict__ geo_w1,
    const float* __restrict__ rw0,    const float* __restrict__ rw1,
    const float* __restrict__ rw2,    const float* __restrict__ rw3,
    const float* __restrict__ rw4,    const float* __restrict__ sw,
    const float* __restrict__ aw,     unsigned int* __restrict__ wpre)
{
  int t = blockIdx.x;
  int lane = threadIdx.x;
  const float* W = nullptr;
  // mode: 0 = plain matrix slice, 1 = combined head, 2 = geo opacity col, 3 = geo_w0
  int KT = 8, Kreal = 256, ldn = 256, coloff = 0, local = t, basedw = 0, mode = 0;
  if (t < 128)      { W = geo_w1; KT = 8; Kreal = 256; ldn = 257; coloff = 1; local = t;       basedw = GEO_DW; }
  else if (t < 272) { W = rw0;    KT = 9; Kreal = 265; ldn = 256; coloff = 0; local = t - 128; basedw = R0_DW; }
  else if (t < 400) { W = rw1;    KT = 8; Kreal = 256; ldn = 256; coloff = 0; local = t - 272; basedw = R1_DW; }
  else if (t < 528) { W = rw2;    KT = 8; Kreal = 256; ldn = 256; coloff = 0; local = t - 400; basedw = R2_DW; }
  else if (t < 656) { W = rw3;    KT = 8; Kreal = 256; ldn = 256; coloff = 0; local = t - 528; basedw = R3_DW; }
  else if (t < 664) { mode = 1;   KT = 8;                                    local = t - 656; basedw = HEAD_DW; }
  else if (t < 672) { mode = 2;   KT = 8;                                    local = t - 664; basedw = GEOOP_DW; }
  else              { mode = 3;   KT = 1;                                    local = t - 672; basedw = GW0_DW; }

  int kt = local % KT;
  int nt = local / KT;
  int n  = nt * 16 + (lane & 15);
  int kh = lane >> 4;
  unsigned int* outp = wpre + basedw + local * 256 + lane * 8;

#pragma unroll
  for (int v = 0; v < 8; ++v) {
    int k0 = kt * 32 + kh * 16 + 2 * v;
    int k1 = k0 + 1;
    float f0, f1;
    if (mode == 0) {
      f0 = (k0 < Kreal) ? W[k0 * ldn + coloff + n] : 0.0f;
      f1 = (k1 < Kreal) ? W[k1 * ldn + coloff + n] : 0.0f;
    } else if (mode == 1) {   // [rw4 | sw | aw | 0]
      f0 = (n < 3) ? rw4[k0 * 3 + n] : (n < 6) ? sw[k0 * 3 + (n - 3)] : (n == 6) ? aw[k0] : 0.0f;
      f1 = (n < 3) ? rw4[k1 * 3 + n] : (n < 6) ? sw[k1 * 3 + (n - 3)] : (n == 6) ? aw[k1] : 0.0f;
    } else if (mode == 2) {   // geo_w1[:,0] in column 0
      f0 = (n == 0) ? geo_w1[k0 * 257] : 0.0f;
      f1 = (n == 0) ? geo_w1[k1 * 257] : 0.0f;
    } else {                  // geo_w0: 3x256, K padded to 32
      f0 = (k0 < 3) ? geo_w0[k0 * 256 + n] : 0.0f;
      f1 = (k1 < 3) ? geo_w0[k1 * 256 + n] : 0.0f;
    }
    outp[v] = pack2(f0, f1);
  }
}

// ---------------------------------------------------------------------------
// One 16xK @ KxN(=256) bf16 WMMA layer; A from LDS, B from prepped global,
// result (+bias, opt ReLU) written back to LDS as bf16 in C/D lane layout.
// ---------------------------------------------------------------------------
template <int KT>
__device__ __forceinline__ void mlp_layer(
    const unsigned short* Ain, int inStride,
    unsigned short* Aout, int outStride, int outColOff,
    const unsigned int* __restrict__ wtiles,
    const float* __restrict__ bias, int biasOff, bool relu, int lane)
{
  int m  = lane & 15;
  int kh = lane >> 4;

  v16bf a[KT];
#pragma unroll
  for (int kt = 0; kt < KT; ++kt) {
    const unsigned short* p = Ain + m * inStride + kt * 32 + kh * 16;
    u32x4 q0 = *(const u32x4*)p;            // K = kt*32 + kh*16 + 0..7
    u32x4 q1 = *(const u32x4*)(p + 8);      // K = ... + 8
    a[kt] = mkv(q0, q1);
  }

#pragma unroll 1
  for (int nt = 0; nt < 16; ++nt) {
    v8f c = {0.f, 0.f, 0.f, 0.f, 0.f, 0.f, 0.f, 0.f};
    const unsigned int* bp = wtiles + nt * KT * 256 + lane * 8;
#pragma unroll
    for (int kt = 0; kt < KT; ++kt) {
      u32x4 b0 = *(const u32x4*)bp;
      u32x4 b1 = *(const u32x4*)(bp + 4);
      c = __builtin_amdgcn_wmma_f32_16x16x32_bf16(false, a[kt], false, mkv(b0, b1),
                                                  (short)0, c, false, false);
      bp += 256;
    }
    int n = nt * 16 + m;
    float bv = bias[biasOff + n];
#pragma unroll
    for (int r = 0; r < 8; ++r) {
      float v = c[r] + bv;
      if (relu) v = fmaxf(v, 0.0f);
      Aout[(r + 8 * kh) * outStride + outColOff + n] = f2bf(v);
    }
  }
}

// head: 16x256 @ 256x16 (cols: rw4[0..2] | sw[0..2] | aw | pad) -> f32 LDS
__device__ __forceinline__ void head_layer(
    const unsigned short* Ain, int inStride,
    const unsigned int* __restrict__ wtiles,
    const float* __restrict__ rb4, const float* __restrict__ sb,
    const float* __restrict__ ab, float* hout, int lane)
{
  int m = lane & 15, kh = lane >> 4;
  v16bf a[8];
#pragma unroll
  for (int kt = 0; kt < 8; ++kt) {
    const unsigned short* p = Ain + m * inStride + kt * 32 + kh * 16;
    a[kt] = mkv(*(const u32x4*)p, *(const u32x4*)(p + 8));
  }
  v8f c = {0.f, 0.f, 0.f, 0.f, 0.f, 0.f, 0.f, 0.f};
  const unsigned int* bp = wtiles + lane * 8;
#pragma unroll
  for (int kt = 0; kt < 8; ++kt) {
    u32x4 b0 = *(const u32x4*)bp;
    u32x4 b1 = *(const u32x4*)(bp + 4);
    c = __builtin_amdgcn_wmma_f32_16x16x32_bf16(false, a[kt], false, mkv(b0, b1),
                                                (short)0, c, false, false);
    bp += 256;
  }
  int n = m;
  float bv = (n < 3) ? rb4[n] : (n < 6) ? sb[n - 3] : (n == 6) ? ab[0] : 0.0f;
#pragma unroll
  for (int r = 0; r < 8; ++r)
    hout[(r + 8 * kh) * 16 + n] = c[r] + bv;
}

// opacity column: gh(16x256) @ geo_w1[:,0] (256x16, col0 only) -> s_op[16]
__device__ __forceinline__ void opacity_wmma(
    const unsigned short* Ain, const unsigned int* __restrict__ wtiles,
    const float* __restrict__ geo_b1, float* s_op, int lane)
{
  int m = lane & 15, kh = lane >> 4;
  v16bf a[8];
#pragma unroll
  for (int kt = 0; kt < 8; ++kt) {
    const unsigned short* p = Ain + m * ASTR + kt * 32 + kh * 16;
    a[kt] = mkv(*(const u32x4*)p, *(const u32x4*)(p + 8));
  }
  v8f c = {0.f, 0.f, 0.f, 0.f, 0.f, 0.f, 0.f, 0.f};
  const unsigned int* bp = wtiles + lane * 8;
#pragma unroll
  for (int kt = 0; kt < 8; ++kt) {
    u32x4 b0 = *(const u32x4*)bp;
    u32x4 b1 = *(const u32x4*)(bp + 4);
    c = __builtin_amdgcn_wmma_f32_16x16x32_bf16(false, a[kt], false, mkv(b0, b1),
                                                (short)0, c, false, false);
    bp += 256;
  }
  if (m == 0) {                 // lanes 0 (rows 0..7) and 16 (rows 8..15) hold col 0
    float bv = geo_b1[0];
#pragma unroll
    for (int r = 0; r < 8; ++r) s_op[r + 8 * kh] = c[r] + bv;
  }
}

// ---------------------------------------------------------------------------
// Main fused kernel: one wave == 16 faces, full pipeline in LDS/VGPRs.
// ---------------------------------------------------------------------------
__global__ __launch_bounds__(64) void mesh_gauss_main(
    const float* __restrict__ vertices, const float* __restrict__ cam,
    const float* __restrict__ geo_b0,   const float* __restrict__ geo_b1,
    const float* __restrict__ rb0, const float* __restrict__ rb1,
    const float* __restrict__ rb2, const float* __restrict__ rb3,
    const float* __restrict__ rb4, const float* __restrict__ sb,
    const float* __restrict__ ab,
    const int* __restrict__ faces, const unsigned int* __restrict__ wpre,
    float* __restrict__ out, int F, int NT)
{
  __shared__ __align__(16) unsigned short s_bufA[2][16 * ASTR];
  __shared__ __align__(16) unsigned short s_bufX[2][16 * XSTR];
  __shared__ float s_head[2][16][16];
  __shared__ float s_op[2][16];

  int w = threadIdx.x >> 5, lane = threadIdx.x & 31;
  int tile = blockIdx.x * 2 + w;
  if (tile >= NT) return;

  unsigned short* A0 = s_bufA[w];
  unsigned short* X  = s_bufX[w];
  float (*hsh)[16]   = s_head[w];
  float* opsh        = s_op[w];

  int m = lane & 15, kh = lane >> 4;
  int f = tile * 16 + m;
  bool valid = (lane < 16) && (f < F);

  float px = 0, py = 0, pz = 0, nx = 0, ny = 0, nz = 0, vdx = 0, vdy = 0, vdz = 0;
  if (valid) {
    int i0 = faces[f * 3 + 0], i1 = faces[f * 3 + 1], i2 = faces[f * 3 + 2];
    float ax = vertices[i0 * 3], ay = vertices[i0 * 3 + 1], az = vertices[i0 * 3 + 2];
    float bx = vertices[i1 * 3], by = vertices[i1 * 3 + 1], bz = vertices[i1 * 3 + 2];
    float cx = vertices[i2 * 3], cy = vertices[i2 * 3 + 1], cz = vertices[i2 * 3 + 2];
    px = (ax + bx + cx) * (1.0f / 3.0f);
    py = (ay + by + cy) * (1.0f / 3.0f);
    pz = (az + bz + cz) * (1.0f / 3.0f);
    float e1x = ax - bx, e1y = ay - by, e1z = az - bz;
    float e2x = ax - cx, e2y = ay - cy, e2z = az - cz;
    nx = e1y * e2z - e1z * e2y;
    ny = e1z * e2x - e1x * e2z;
    nz = e1x * e2y - e1y * e2x;
    float nn = fmaxf(sqrtf(nx * nx + ny * ny + nz * nz), 1e-12f);
    nx /= nn; ny /= nn; nz /= nn;
    vdx = px - cam[0]; vdy = py - cam[1]; vdz = pz - cam[2];
    float vn = fmaxf(sqrtf(vdx * vdx + vdy * vdy + vdz * vdz), 1e-12f);
    vdx /= vn; vdy /= vn; vdz /= vn;
  }

  // --- gh = softplus(xyz @ geo_w0 + b0) via WMMA -------------------------
  // A tile (16x32): row m = lane&15; kh==0 lanes hold K=0..2 = xyz, rest 0.
  v16bf axyz;
  {
    union { unsigned short s[16]; v16bf v; } ua;
#pragma unroll
    for (int i = 0; i < 16; ++i) ua.s[i] = 0;
    if (kh == 0) { ua.s[0] = f2bf(px); ua.s[1] = f2bf(py); ua.s[2] = f2bf(pz); }
    axyz = ua.v;
  }
#pragma unroll 1
  for (int nt = 0; nt < 16; ++nt) {
    const unsigned int* bp = wpre + GW0_DW + nt * 256 + lane * 8;
    u32x4 b0 = *(const u32x4*)bp;
    u32x4 b1 = *(const u32x4*)(bp + 4);
    v8f c = {0.f, 0.f, 0.f, 0.f, 0.f, 0.f, 0.f, 0.f};
    c = __builtin_amdgcn_wmma_f32_16x16x32_bf16(false, axyz, false, mkv(b0, b1),
                                                (short)0, c, false, false);
    int n = nt * 16 + m;
    float bv = geo_b0[n];
#pragma unroll
    for (int r = 0; r < 8; ++r)
      A0[(r + 8 * kh) * ASTR + n] = f2bf(softplusf(c[r] + bv));
  }
  lds_fence();

  // opacity_logit == raw geo output column 0 (logit(sigmoid(x)) == x)
  opacity_wmma(A0, wpre + GEOOP_DW, geo_b1, opsh, lane);

  // feat = gh @ geo_w1[:,1:] + b1[1:]  -> X cols 9..264 (linear)
  mlp_layer<8>(A0, ASTR, X, XSTR, 9, wpre + GEO_DW, geo_b1, 1, false, lane);

  // X cols 0..8 = [xyz, view_dir, normal]; zero the K-pad cols 265..287
  if (lane < 16) {
    unsigned short* xr = X + m * XSTR;
    xr[0] = f2bf(px);  xr[1] = f2bf(py);  xr[2] = f2bf(pz);
    xr[3] = f2bf(vdx); xr[4] = f2bf(vdy); xr[5] = f2bf(vdz);
    xr[6] = f2bf(nx);  xr[7] = f2bf(ny);  xr[8] = f2bf(nz);
  }
  for (int idx = lane; idx < 16 * 23; idx += 32) {
    int rr = idx / 23, cc = 265 + idx % 23;
    X[rr * XSTR + cc] = 0;
  }
  lds_fence();

  mlp_layer<9>(X,  XSTR, A0, ASTR, 0, wpre + R0_DW, rb0, 0, true, lane); lds_fence();
  mlp_layer<8>(A0, ASTR, X,  XSTR, 0, wpre + R1_DW, rb1, 0, true, lane); lds_fence();
  mlp_layer<8>(X,  XSTR, A0, ASTR, 0, wpre + R2_DW, rb2, 0, true, lane); lds_fence();
  mlp_layer<8>(A0, ASTR, X,  XSTR, 0, wpre + R3_DW, rb3, 0, true, lane); lds_fence();

  head_layer(X, XSTR, wpre + HEAD_DW, rb4, sb, ab, &hsh[0][0], lane);
  lds_fence();

  if (valid) {
    float op_logit = opsh[m];
    float c0 = sigmoidf(hsh[m][0]), c1 = sigmoidf(hsh[m][1]), c2 = sigmoidf(hsh[m][2]);
    float s0 = sigmoidf(hsh[m][3]), s1 = sigmoidf(hsh[m][4]), s2 = sigmoidf(hsh[m][5]);
    float th = sigmoidf(hsh[m][6]) * 6.283185307179586f;
    float fd0 = (c0 - 0.5f) / C0F, fd1 = (c1 - 0.5f) / C0F, fd2 = (c2 - 0.5f) / C0F;
    float sl0 = __logf(s0), sl1 = __logf(s1), sl2 = __logf(s2);
    float hh = 0.5f * th;
    float qr = __cosf(hh), sh = __sinf(hh);
    float qx = sh * nx, qy = sh * ny, qz = sh * nz;
    float qn = fmaxf(sqrtf(qr * qr + qx * qx + qy * qy + qz * qz), 1e-12f);
    float r = qr / qn, x_ = qx / qn, y_ = qy / qn, z_ = qz / qn;
    float R00 = 1.f - 2.f * (y_ * y_ + z_ * z_), R01 = 2.f * (x_ * y_ - r * z_), R02 = 2.f * (x_ * z_ + r * y_);
    float R10 = 2.f * (x_ * y_ + r * z_), R11 = 1.f - 2.f * (x_ * x_ + z_ * z_), R12 = 2.f * (y_ * z_ - r * x_);
    float R20 = 2.f * (x_ * z_ - r * y_), R21 = 2.f * (y_ * z_ + r * x_), R22 = 1.f - 2.f * (x_ * x_ + y_ * y_);
    float L00 = R00 * s0, L01 = R01 * s1, L02 = R02 * s2;
    float L10 = R10 * s0, L11 = R11 * s1, L12 = R12 * s2;
    float L20 = R20 * s0, L21 = R21 * s1, L22 = R22 * s2;
    float cov00 = L00 * L00 + L01 * L01 + L02 * L02;
    float cov01 = L00 * L10 + L01 * L11 + L02 * L12;
    float cov02 = L00 * L20 + L01 * L21 + L02 * L22;
    float cov11 = L10 * L10 + L11 * L11 + L12 * L12;
    float cov12 = L10 * L20 + L11 * L21 + L12 * L22;
    float cov22 = L20 * L20 + L21 * L21 + L22 * L22;

    float* o = out + (size_t)f * 23;
    o[0]  = px;  o[1]  = py;  o[2]  = pz;
    o[3]  = nx;  o[4]  = ny;  o[5]  = nz;
    o[6]  = fd0; o[7]  = fd1; o[8]  = fd2;
    o[9]  = sl0; o[10] = sl1; o[11] = sl2;
    o[12] = qr;  o[13] = qx;  o[14] = qy;  o[15] = qz;   // unnormalized quat (as reference)
    o[16] = op_logit;
    o[17] = cov00; o[18] = cov01; o[19] = cov02;
    o[20] = cov11; o[21] = cov12; o[22] = cov22;
  }
}

extern "C" void kernel_launch(void* const* d_in, const int* in_sizes, int n_in,
                              void* d_out, int out_size, void* d_ws, size_t ws_size,
                              hipStream_t stream)
{
  (void)n_in; (void)out_size; (void)ws_size;
  const float* vertices = (const float*)d_in[0];
  const float* cam      = (const float*)d_in[1];
  const float* geo_w0   = (const float*)d_in[2];
  const float* geo_b0   = (const float*)d_in[3];
  const float* geo_w1   = (const float*)d_in[4];
  const float* geo_b1   = (const float*)d_in[5];
  const float* rw0      = (const float*)d_in[6];
  const float* rb0      = (const float*)d_in[7];
  const float* rw1      = (const float*)d_in[8];
  const float* rb1      = (const float*)d_in[9];
  const float* rw2      = (const float*)d_in[10];
  const float* rb2      = (const float*)d_in[11];
  const float* rw3      = (const float*)d_in[12];
  const float* rb3      = (const float*)d_in[13];
  const float* rw4      = (const float*)d_in[14];
  const float* rb4      = (const float*)d_in[15];
  const float* sw       = (const float*)d_in[16];
  const float* sb       = (const float*)d_in[17];
  const float* aw       = (const float*)d_in[18];
  const float* ab       = (const float*)d_in[19];
  const int*   faces    = (const int*)d_in[20];

  int F = in_sizes[20] / 3;
  unsigned int* wpre = (unsigned int*)d_ws;

  prep_weights<<<dim3(NPREP_TILES), dim3(32), 0, stream>>>(
      geo_w0, geo_w1, rw0, rw1, rw2, rw3, rw4, sw, aw, wpre);

  int NT = (F + 15) / 16;
  int nb = (NT + 1) / 2;   // 2 waves (tiles) per 64-thread block
  mesh_gauss_main<<<dim3(nb), dim3(64), 0, stream>>>(
      vertices, cam, geo_b0, geo_b1,
      rb0, rb1, rb2, rb3, rb4, sb, ab,
      faces, wpre, (float*)d_out, F, NT);
}
